// Res2D_35588099014803
// MI455X (gfx1250) — matrix operand order repaired and checked by reading, not compiled, not run on hardware
//
#include <hip/hip_runtime.h>
#include <hip/hip_bf16.h>
#include <math.h>

// ---------------------------------------------------------------------------
// Res2D reservoir on MI455X (gfx1250, wave32).
//   embed GEMM + per-step readout GEMM -> v_wmma_f32_16x16x32_bf16 (f32 acc)
//   depthwise circular stencils        -> fp32 VALU on 40x40 LDS tiles
//   deconv (stride==kernel)            -> fused 8-fma per pixel in step kernel
// Everything L2-resident (< 8 MB workspace); scan serialized via per-step
// launches (graph-captured by the harness).
// ---------------------------------------------------------------------------

typedef __attribute__((ext_vector_type(16))) __bf16 v16bf;
typedef __attribute__((ext_vector_type(8)))  __bf16 v8bf;
typedef __attribute__((ext_vector_type(8)))  float  v8f;
typedef __attribute__((ext_vector_type(4)))  float  v4f;

#define DIM      256
#define NCH      8
#define TSTEPS   128
#define EMB_N    2048   // C_INT*ISQ*ISQ
#define EMB_K    512    // INPUT_DIM
#define RO_K     8192   // C * 32 * 32
#define RO_N     16     // OUT_CH

// -------------------------------- utility ----------------------------------

__global__ void cvt_bf16_kernel(const float* __restrict__ in,
                                __bf16* __restrict__ out, int n) {
  int i = blockIdx.x * 256 + threadIdx.x;
  if (i < n) out[i] = (__bf16)in[i];
}

__global__ void zero_kernel(float* __restrict__ p, int n) {
  int i = blockIdx.x * 256 + threadIdx.x;
  if (i < n) p[i] = 0.0f;
}

// ------------------------------ embed GEMM ---------------------------------
// ue[m,n] = mask_coarse[n & 255] * sum_k X[m,k] * W_embed[n,k]
// M=128, N=2048, K=512.  One 16x16 tile per wave, 4 waves per block.
__global__ void embed_gemm_kernel(const __bf16* __restrict__ Xbf,   // [128][512]
                                  const __bf16* __restrict__ Wbf,   // [2048][512]
                                  const float*  __restrict__ mask_coarse,
                                  float* __restrict__ ue) {         // [128][2048]
  const int tid  = threadIdx.x;
  const int lane = tid & 31;
  const int wave = tid >> 5;
  const int tile = blockIdx.x * 4 + wave;      // 0..1023
  const int mtile = tile >> 7;                 // 0..7   (128/16)
  const int ntile = tile & 127;                // 0..127 (2048/16)
  const int half  = (lane >> 4) & 1;
  const int lr    = lane & 15;

  const __bf16* arow = Xbf + (mtile * 16 + lr) * EMB_K;
  const __bf16* brow = Wbf + (ntile * 16 + lr) * EMB_K;

  v8f acc = {};
  #pragma unroll 4
  for (int kb = 0; kb < EMB_K; kb += 32) {
    union { v16bf v; v8bf h[2]; } a;
    a.h[0] = *(const v8bf*)(arow + kb + half * 16);       // K  kb+h*16 .. +7
    a.h[1] = *(const v8bf*)(arow + kb + half * 16 + 8);   // K  kb+h*16+8 ..
    v16bf b = *(const v16bf*)(brow + kb + half * 16);     // K  kb+h*16 .. +15
    acc = __builtin_amdgcn_wmma_f32_16x16x32_bf16(
        false, a.v, false, b, (short)0, acc, false, false);
  }

  const int n = ntile * 16 + lr;
  const float msk = mask_coarse[n & 255];   // n = cint*256 + i*16 + j
  #pragma unroll
  for (int r = 0; r < 8; ++r) {
    int m = mtile * 16 + half * 8 + r;      // D: VGPR r -> M = r + 8*half
    ue[m * EMB_N + n] = acc[r] * msk;
  }
}

// ------------------------------ scan step ----------------------------------
// Z_new = tanh(0.9*conv3x3_d1(Z) + 0.1*conv5x5_d2(Z) + mask_fine * deconv(ue_t))
// One 32x32 tile of one channel per 256-thread block; LDS tile 40x40 (halo 4).
__global__ void step_kernel(const float* __restrict__ Zin,
                            float* __restrict__ Zout,
                            const float* __restrict__ ue_t,     // [8][16][16]
                            const float* __restrict__ Wd,       // [8][8][16][16]
                            const float* __restrict__ w1,       // [8][3][3]
                            const float* __restrict__ w2,       // [8][5][5]
                            const float* __restrict__ mask_fine) {
  __shared__ float tile[40 * 40];
  __shared__ float wa[9];
  __shared__ float wb[25];

  const int c   = blockIdx.z;
  const int ty0 = blockIdx.y * 32;
  const int tx0 = blockIdx.x * 32;
  const int tid = threadIdx.x;
  const float* zc = Zin + c * (DIM * DIM);

  for (int idx = tid; idx < 40 * 40; idx += 256) {
    int ly = idx / 40, lx = idx - ly * 40;
    int gy = (ty0 - 4 + ly) & (DIM - 1);      // circular padding
    int gx = (tx0 - 4 + lx) & (DIM - 1);
    tile[idx] = zc[gy * DIM + gx];
  }
  if (tid < 9) wa[tid] = w1[c * 9 + tid];
  if (tid >= 32 && tid < 57) wb[tid - 32] = w2[c * 25 + tid - 32];
  __syncthreads();

  #pragma unroll
  for (int q = 0; q < 4; ++q) {
    int o  = tid + q * 256;
    int ly = o >> 5, lx = o & 31;

    float s3 = 0.0f;                           // 3x3, dilation 1
    #pragma unroll
    for (int dy = 0; dy < 3; ++dy)
      #pragma unroll
      for (int dx = 0; dx < 3; ++dx)
        s3 = fmaf(wa[dy * 3 + dx], tile[(ly + 3 + dy) * 40 + (lx + 3 + dx)], s3);

    float s5 = 0.0f;                           // 5x5, dilation 2
    #pragma unroll
    for (int dy = 0; dy < 5; ++dy)
      #pragma unroll
      for (int dx = 0; dx < 5; ++dx)
        s5 = fmaf(wb[dy * 5 + dx], tile[(ly + 2 * dy) * 40 + (lx + 2 * dx)], s5);

    int gy = ty0 + ly, gx = tx0 + lx;
    int bi = gy >> 4, a = gy & 15, bj = gx >> 4, b = gx & 15;
    float xin = 0.0f;                          // fused block-deconv (8 fmas)
    #pragma unroll
    for (int ci = 0; ci < 8; ++ci)
      xin = fmaf(ue_t[ci * 256 + bi * 16 + bj],
                 Wd[(ci * 8 + c) * 256 + a * 16 + b], xin);
    xin *= mask_fine[gy * DIM + gx];

    Zout[c * (DIM * DIM) + gy * DIM + gx] =
        tanhf(fmaf(0.9f, s3, fmaf(0.1f, s5, xin)));
  }
}

// ------------------------------ readout ------------------------------------
// y[n, m] = b_out[n] + sum_k patch[m,k] * w_out[n,k]; M=64 blocks, N=16, K=8192
// k = c*1024 + a*32 + b ; patch row m=(bi*8+bj) reads Z[c, bi*32+a, bj*32+b].
// 512 threads = 16 waves: wave = (kchunk 0..3) x (mtile 0..3).  Each wave does
// a 16x16 D tile over K=2048, partials combined deterministically through LDS.
__global__ void readout_kernel(const float* __restrict__ Z,      // [8][256][256]
                               const __bf16* __restrict__ WrT,   // [16][8192]
                               const float* __restrict__ b_out,  // [16]
                               float* __restrict__ y) {          // [16][64]
  __shared__ float red[4 * 1024];                // [kchunk][oc*64 + block]

  const int tid   = threadIdx.x;                 // 0..511
  const int lane  = tid & 31;
  const int wave  = tid >> 5;                    // 0..15
  const int mtile = wave & 3;
  const int kc    = wave >> 2;                   // K chunk 0..3 (2048 each)
  const int half  = (lane >> 4) & 1;
  const int lr    = lane & 15;
  const int m     = mtile * 16 + lr;             // A row = block index 0..63
  const int bi    = m >> 3, bj = m & 7;
  const __bf16* brow = WrT + lr * RO_K;

  v8f acc = {};
  const int kbeg = kc * 2048;
  #pragma unroll 4
  for (int kk = 0; kk < 2048; kk += 32) {
    int kb = kbeg + kk;
    int k0 = kb + half * 16;                     // elems 0..7
    int c0 = k0 >> 10, a0 = (k0 >> 5) & 31, b0 = k0 & 31;
    const float* p0 = Z + c0 * (DIM * DIM) + (bi * 32 + a0) * DIM + bj * 32 + b0;
    int k1 = k0 + 8;                             // elems 8..15
    int c1 = k1 >> 10, a1 = (k1 >> 5) & 31, b1 = k1 & 31;
    const float* p1 = Z + c1 * (DIM * DIM) + (bi * 32 + a1) * DIM + bj * 32 + b1;

    v4f f0 = *(const v4f*)p0;
    v4f f1 = *(const v4f*)(p0 + 4);
    v4f f2 = *(const v4f*)p1;
    v4f f3 = *(const v4f*)(p1 + 4);

    union { v16bf v; __bf16 e[16]; } a;
    #pragma unroll
    for (int e = 0; e < 4; ++e) {
      a.e[e]      = (__bf16)f0[e];
      a.e[4 + e]  = (__bf16)f1[e];
      a.e[8 + e]  = (__bf16)f2[e];
      a.e[12 + e] = (__bf16)f3[e];
    }
    v16bf b = *(const v16bf*)(brow + kb + half * 16);
    acc = __builtin_amdgcn_wmma_f32_16x16x32_bf16(
        false, a.v, false, b, (short)0, acc, false, false);
  }

  #pragma unroll
  for (int r = 0; r < 8; ++r) {
    int mm = mtile * 16 + half * 8 + r;          // block index
    red[kc * 1024 + lr * 64 + mm] = acc[r];      // partial at (oc=lr, block=mm)
  }
  __syncthreads();

  // deterministic combine of the 4 K-chunk partials + bias
  for (int o = tid; o < 1024; o += 512) {
    int n = o >> 6;                              // oc
    y[o] = b_out[n] + ((red[o] + red[1024 + o]) + (red[2048 + o] + red[3072 + o]));
  }
}

// ------------------------------ launcher -----------------------------------

// workspace layout (bytes, 256-aligned)
#define OFF_XBF  0u          // 128*512  bf16 = 131072
#define OFF_WBF  131072u     // 2048*512 bf16 = 2097152
#define OFF_WRT  2228224u    // 16*8192  bf16 = 262144
#define OFF_UE   2490368u    // 128*2048 f32  = 1048576
#define OFF_Z0   3538944u    // 8*256*256 f32 = 2097152
#define OFF_Z1   5636096u    // 8*256*256 f32 = 2097152

extern "C" void kernel_launch(void* const* d_in, const int* in_sizes, int n_in,
                              void* d_out, int out_size, void* d_ws, size_t ws_size,
                              hipStream_t stream) {
  const float* X           = (const float*)d_in[0];   // [128,512]
  const float* W_embed     = (const float*)d_in[1];   // [2048,512]
  const float* mask_coarse = (const float*)d_in[2];   // [16,16]
  const float* mask_fine   = (const float*)d_in[3];   // [256,256]
  const float* W_deconv    = (const float*)d_in[4];   // [8,8,16,16]
  const float* w1          = (const float*)d_in[5];   // [8,1,3,3]
  const float* w2          = (const float*)d_in[6];   // [8,1,5,5]
  const float* w_out       = (const float*)d_in[7];   // [16,8,32,32]
  const float* b_out       = (const float*)d_in[8];   // [16]
  float* out = (float*)d_out;                         // [128,1024]

  char* ws = (char*)d_ws;
  __bf16* Xbf = (__bf16*)(ws + OFF_XBF);
  __bf16* Wbf = (__bf16*)(ws + OFF_WBF);
  __bf16* WrT = (__bf16*)(ws + OFF_WRT);
  float*  ue  = (float*) (ws + OFF_UE);
  float*  Z0  = (float*) (ws + OFF_Z0);
  float*  Z1  = (float*) (ws + OFF_Z1);

  // precision staging: bf16 operands for the two WMMA GEMMs
  cvt_bf16_kernel<<<(128 * EMB_K + 255) / 256, 256, 0, stream>>>(X, Xbf, 128 * EMB_K);
  cvt_bf16_kernel<<<(EMB_N * EMB_K + 255) / 256, 256, 0, stream>>>(W_embed, Wbf, EMB_N * EMB_K);
  // w_out [oc][c][32][32] flattened IS B^T[n][k] with k=(c,a,b): plain convert
  cvt_bf16_kernel<<<(RO_N * RO_K + 255) / 256, 256, 0, stream>>>(w_out, WrT, RO_N * RO_K);
  zero_kernel<<<(NCH * DIM * DIM + 255) / 256, 256, 0, stream>>>(Z0, NCH * DIM * DIM);

  // embed: 1024 wave tiles, 4 waves / 128-thread block
  embed_gemm_kernel<<<256, 128, 0, stream>>>(Xbf, Wbf, mask_coarse, ue);

  // serial reservoir scan: ping-pong state, fused deconv + stencils + tanh,
  // then WMMA readout of this timestep (16 waves, K split 4 ways)
  float* zin = Z0;
  float* zout = Z1;
  for (int t = 0; t < TSTEPS; ++t) {
    step_kernel<<<dim3(8, 8, 8), 256, 0, stream>>>(
        zin, zout, ue + t * EMB_N, W_deconv, w1, w2, mask_fine);
    readout_kernel<<<1, 512, 0, stream>>>(zout, WrT, b_out, out + t * 1024);
    float* tmp = zin; zin = zout; zout = tmp;
  }
  (void)in_sizes; (void)n_in; (void)out_size; (void)ws_size;
}
